// Shape2DPainterNet_38938173505671
// MI455X (gfx1250) — matrix operand order repaired and checked by reading, not compiled, not run on hardware
//
#include <hip/hip_runtime.h>
#include <hip/hip_bf16.h>
#include <math.h>

// ---------------------------------------------------------------------------
// Shape2DPainterNet for MI455X (gfx1250, wave32, WMMA).
// LSTM recurrence runs on v_wmma_f32_16x16x32_bf16 with weights staged in LDS.
// ---------------------------------------------------------------------------

typedef __attribute__((ext_vector_type(16))) __bf16 v16bf;
typedef __attribute__((ext_vector_type(8)))  float  v8f;

#define RNN   64
#define SEQ   32
#define GATES 256   // 4 * RNN

// ---------------- math helpers ----------------
__device__ __forceinline__ float sigf(float x) {
  return 1.0f / (1.0f + __expf(-x));
}
__device__ __forceinline__ float tanh_fast(float x) {
  return 2.0f / (1.0f + __expf(-2.0f * x)) - 1.0f;
}

// ---------------- WMMA operand loaders (contiguous k = 16*half + e in both A and B) -------------
// A-matrix 16x32 bf16: lane L (half = L>>4, m = L&15) holds row m;
//   element e: K = half*16 + e (matches B layout).
__device__ __forceinline__ v16bf load_A16(const __bf16* m, int row0, int kc, int lane) {
  const int half = lane >> 4;
  const __bf16* p = m + (row0 + (lane & 15)) * RNN + kc * 32 + half * 16;
  v16bf a;
#pragma unroll
  for (int v = 0; v < 8; ++v) {
    const int kb = v * 2;
    a[2 * v]     = p[kb];
    a[2 * v + 1] = p[kb + 1];
  }
  return a;
}

// B-matrix 32x16 bf16: lane L holds column n = L&15; element e: K = (L>>4)*16 + e.
// W is row-major [256 x 64]; B[k][n] = W[n][k] (gates = x @ W^T).
__device__ __forceinline__ v16bf load_B16(const __bf16* w, int n0, int kc, int lane) {
  const __bf16* p = w + (n0 + (lane & 15)) * RNN + kc * 32 + (lane >> 4) * 16;
  v16bf b;
#pragma unroll
  for (int e = 0; e < 16; ++e) b[e] = p[e];
  return b;
}

#define WMMA_BF16(A, B, C) \
  __builtin_amdgcn_wmma_f32_16x16x32_bf16(false, (A), false, (B), (short)0, (C), false, false)

// ---------------------------------------------------------------------------
// Kernel 1: full 32-step LSTM. blockDim = 128 (4 waves); each wave owns a
// 16-row batch tile; block covers 64 batch rows. gridDim.y selects dialog step.
// ---------------------------------------------------------------------------
__global__ void __launch_bounds__(128)
lstm_kernel(const int* __restrict__ inst0, const int* __restrict__ inst1,
            const float* __restrict__ embed,
            const float* __restrict__ W_ih, const float* __restrict__ W_hh,
            const float* __restrict__ b_ih, const float* __restrict__ b_hh,
            float* __restrict__ h0_out, float* __restrict__ h1_out, int B) {
  extern __shared__ char smem[];
  __bf16* wih  = (__bf16*)smem;                    // 256*64 bf16 = 32 KB
  __bf16* whh  = wih + GATES * RNN;                // 32 KB
  float*  bias = (float*)(whh + GATES * RNN);      // 256 f32
  __bf16* xls  = (__bf16*)(bias + GATES);          // 64*64 bf16 = 8 KB
  __bf16* hls  = xls + 64 * RNN;                   // 8 KB

  const int tid  = threadIdx.x;
  const int lane = tid & 31;
  const int wv   = tid >> 5;
  const int b0   = blockIdx.x * 64;
  const int* inst = blockIdx.y ? inst1 : inst0;
  float* hout     = blockIdx.y ? h1_out : h0_out;

  // Stage weights once (f32 -> bf16) and zero h_{-1}.
  for (int i = tid; i < GATES * RNN; i += 128) {
    wih[i] = (__bf16)W_ih[i];
    whh[i] = (__bf16)W_hh[i];
  }
  for (int i = tid; i < GATES; i += 128) bias[i] = b_ih[i] + b_hh[i];
  for (int i = tid; i < 64 * RNN; i += 128) hls[i] = (__bf16)0.0f;
  __syncthreads();

  // Cell state c[16 rows x 64 cols] per wave, held in accumulator layout:
  // element (r, lane) of group g -> row 16*wv + r + 8*(lane>>4), col g*16 + (lane&15)
  v8f c_st[4];
#pragma unroll
  for (int g = 0; g < 4; ++g)
#pragma unroll
    for (int r = 0; r < 8; ++r) c_st[g][r] = 0.0f;

  const int half = lane >> 4;
  const int nlo  = lane & 15;

#pragma unroll 1
  for (int t = 0; t < SEQ; ++t) {
    // Gather x_t = embed[inst[:, t]] into LDS (bf16). tid>>1 maps wave wv to
    // rows 16*wv..16*wv+15, so every wave produces exactly the rows it consumes:
    // no block barrier needed, only wave-internal DScnt ordering.
    {
      const int row = tid >> 1;
      const int cb  = (tid & 1) * 32;
      const int tok = inst[(b0 + row) * SEQ + t];
      const float* er = embed + (size_t)tok * RNN + cb;
      __bf16* xr = xls + row * RNN + cb;
#pragma unroll
      for (int c = 0; c < 32; ++c) xr[c] = (__bf16)er[c];
    }

    // Load all A operands for h_{t-1} BEFORE overwriting hls below.
    v16bf ax0 = load_A16(xls, 16 * wv, 0, lane);
    v16bf ax1 = load_A16(xls, 16 * wv, 1, lane);
    v16bf ah0 = load_A16(hls, 16 * wv, 0, lane);
    v16bf ah1 = load_A16(hls, 16 * wv, 1, lane);

    // Gate columns processed in 4 groups of 16; i/f/g/o tiles share (r,lane)
    // positions so the cell update is a pure register operation.
    for (int ntc = 0; ntc < 4; ++ntc) {
      const int n = ntc * 16;
      v8f ai, af, ag, ao;
      const float bi  = bias[0 * RNN + n + nlo];
      const float bff = bias[1 * RNN + n + nlo];
      const float bg  = bias[2 * RNN + n + nlo];
      const float bo  = bias[3 * RNN + n + nlo];
#pragma unroll
      for (int r = 0; r < 8; ++r) { ai[r] = bi; af[r] = bff; ag[r] = bg; ao[r] = bo; }

      ai = WMMA_BF16(ax0, load_B16(wih, 0 * RNN + n, 0, lane), ai);
      ai = WMMA_BF16(ax1, load_B16(wih, 0 * RNN + n, 1, lane), ai);
      ai = WMMA_BF16(ah0, load_B16(whh, 0 * RNN + n, 0, lane), ai);
      ai = WMMA_BF16(ah1, load_B16(whh, 0 * RNN + n, 1, lane), ai);

      af = WMMA_BF16(ax0, load_B16(wih, 1 * RNN + n, 0, lane), af);
      af = WMMA_BF16(ax1, load_B16(wih, 1 * RNN + n, 1, lane), af);
      af = WMMA_BF16(ah0, load_B16(whh, 1 * RNN + n, 0, lane), af);
      af = WMMA_BF16(ah1, load_B16(whh, 1 * RNN + n, 1, lane), af);

      ag = WMMA_BF16(ax0, load_B16(wih, 2 * RNN + n, 0, lane), ag);
      ag = WMMA_BF16(ax1, load_B16(wih, 2 * RNN + n, 1, lane), ag);
      ag = WMMA_BF16(ah0, load_B16(whh, 2 * RNN + n, 0, lane), ag);
      ag = WMMA_BF16(ah1, load_B16(whh, 2 * RNN + n, 1, lane), ag);

      ao = WMMA_BF16(ax0, load_B16(wih, 3 * RNN + n, 0, lane), ao);
      ao = WMMA_BF16(ax1, load_B16(wih, 3 * RNN + n, 1, lane), ao);
      ao = WMMA_BF16(ah0, load_B16(whh, 3 * RNN + n, 0, lane), ao);
      ao = WMMA_BF16(ah1, load_B16(whh, 3 * RNN + n, 1, lane), ao);

#pragma unroll
      for (int r = 0; r < 8; ++r) {
        const float ig = sigf(ai[r]);
        const float fg = sigf(af[r]);
        const float gg = tanh_fast(ag[r]);
        const float og = sigf(ao[r]);
        const float c  = fg * c_st[ntc][r] + ig * gg;
        c_st[ntc][r]   = c;
        const float h  = og * tanh_fast(c);
        const int row  = 16 * wv + r + 8 * half;
        const int col  = n + nlo;
        hls[row * RNN + col] = (__bf16)h;
        if (t == SEQ - 1) hout[(size_t)(b0 + row) * RNN + col] = h;
      }
    }
  }
}

// ---------------------------------------------------------------------------
// Kernel 2: attention MLP  relu([h1,prev] @ Wr1^T + br1) @ Wr2^T + br2
// ---------------------------------------------------------------------------
__global__ void __launch_bounds__(256)
att_kernel(const float* __restrict__ h1, const int* __restrict__ canvas0,
           const float* __restrict__ Wr1, const float* __restrict__ br1,
           const float* __restrict__ Wr2, const float* __restrict__ br2,
           float* __restrict__ att, int B) {
  const int idx = blockIdx.x * blockDim.x + threadIdx.x;
  if (idx >= B * 25) return;
  const int b = idx / 25, cell = idx % 25;
  float feat[68];
  const float* hb = h1 + (size_t)b * RNN;
#pragma unroll
  for (int k = 0; k < 64; ++k) feat[k] = hb[k];
#pragma unroll
  for (int k = 0; k < 4; ++k) feat[64 + k] = (float)canvas0[(size_t)(b * 25 + cell) * 4 + k];
  float out = br2[0];
#pragma unroll 4
  for (int j = 0; j < 32; ++j) {
    float acc = br1[j];
    const float* wr = Wr1 + j * 68;
#pragma unroll
    for (int k = 0; k < 68; ++k) acc += wr[k] * feat[k];
    out += Wr2[j] * fmaxf(acc, 0.0f);
  }
  att[idx] = out;
}

// ---------------------------------------------------------------------------
// Kernel 3: heads + categorical sampling + rewards
// ---------------------------------------------------------------------------
__device__ __forceinline__ float urnd(unsigned j, unsigned b) {
  unsigned s = (j + 1u) * 0x9E3779B9u ^ (b * 0x85EBCA6Bu + 0xC2B2AE35u);
  s ^= s >> 16; s *= 0x7FEB352Du; s ^= s >> 15; s *= 0x846CA68Bu; s ^= s >> 16;
  return (float)(s >> 8) * (1.0f / 16777216.0f);
}

__device__ __forceinline__ int sample_cat(const float* lg, int n, unsigned key,
                                          unsigned b, float* lp) {
  float m = lg[0];
  for (int i = 1; i < n; ++i) m = fmaxf(m, lg[i]);
  float sum = 0.0f;
  for (int i = 0; i < n; ++i) sum += __expf(lg[i] - m);
  const float u = urnd(key, b) * sum;
  float acc = 0.0f;
  int s = n - 1;
  for (int i = 0; i < n; ++i) {
    acc += __expf(lg[i] - m);
    if (u < acc) { s = i; break; }
  }
  *lp = lg[s] - m - __logf(sum);
  return s;
}

template <int N>
__device__ __forceinline__ void head(const float* h, const float* __restrict__ W,
                                     const float* __restrict__ bv, float* out) {
  for (int i = 0; i < N; ++i) {
    float acc = bv[i];
    const float* wr = W + i * RNN;
#pragma unroll
    for (int k = 0; k < RNN; ++k) acc += wr[k] * h[k];
    out[i] = acc;
  }
}

__device__ __forceinline__ void step_rewards(const int* cv1b, int loc, int cs,
                                             float* loc_r, float* col_r) {
  const bool valid = (loc >= 0) && (loc < 25);
  const int lc = min(max(loc, 0), 24);
  const int p0 = cv1b[lc * 4 + 0];
  const int sum = p0 + cv1b[lc * 4 + 1] + cv1b[lc * 4 + 2] + cv1b[lc * 4 + 3];
  const bool ok = valid && (sum >= 0);
  *loc_r = ok ? 1.0f : -1.0f;
  *col_r = ok ? ((cs == p0) ? 1.0f : -1.0f) : 0.0f;
}

__global__ void __launch_bounds__(256)
finalize_kernel(const float* __restrict__ h0, const float* __restrict__ h1,
                const float* __restrict__ att,
                const int* __restrict__ canvas0, const int* __restrict__ canvas1,
                const int* __restrict__ ref,
                const float* __restrict__ Wc, const float* __restrict__ bc,
                const float* __restrict__ Ws, const float* __restrict__ bs,
                const float* __restrict__ Wl, const float* __restrict__ bl,
                const float* __restrict__ Wp, const float* __restrict__ bp,
                float* __restrict__ out, int B) {
  const int b = blockIdx.x * blockDim.x + threadIdx.x;
  if (b >= B) return;

  float hb[RNN];
  float lg[25];
  const float* h0b = h0 + (size_t)b * RNN;
#pragma unroll
  for (int k = 0; k < RNN; ++k) hb[k] = h0b[k];

  float clp0, slp0, llp0;
  head<3>(hb, Wc, bc, lg);  const int cs0  = sample_cat(lg, 3, 0, b, &clp0);
  head<3>(hb, Ws, bs, lg);  const int ss0  = sample_cat(lg, 3, 1, b, &slp0);
  head<25>(hb, Wl, bl, lg); const int loc0 = sample_cat(lg, 25, 2, b, &llp0);
  (void)ss0;
  const int* cv1 = canvas1 + (size_t)b * 100;
  float lr0, cr0; step_rewards(cv1, loc0, cs0, &lr0, &cr0);

  const float* h1b = h1 + (size_t)b * RNN;
#pragma unroll
  for (int k = 0; k < RNN; ++k) hb[k] = h1b[k];

  float clp1, slp1, llp1, alp1;
  head<3>(hb, Wc, bc, lg); const int cs1 = sample_cat(lg, 3, 3, b, &clp1);
  head<3>(hb, Ws, bs, lg); const int ss1 = sample_cat(lg, 3, 4, b, &slp1);
  (void)ss1;
  head<8>(hb, Wp, bp, lg); const int ls1 = sample_cat(lg, 8, 5, b, &llp1);

  for (int i = 0; i < 25; ++i) lg[i] = att[(size_t)b * 25 + i];
  const int as = sample_cat(lg, 25, 6, b, &alp1);

  const int* cv0 = canvas0 + (size_t)b * 100;
  const int r0 = cv0[as * 4 + 0], r1 = cv0[as * 4 + 1];
  const int r2 = cv0[as * 4 + 2], r3 = cv0[as * 4 + 3];
  const int* rf = ref + (size_t)b * 4;
  const bool match = (r0 == rf[0]) && (r1 == rf[1]) && (r2 == rf[2]) && (r3 == rf[3]);
  const float att_r = match ? 1.0f : -1.0f;

  const int offx[8] = {-1, 1, 0, 0, -1, -1, 1, 1};
  const int offy[8] = { 0, 0, 1, -1, -1,  1, -1, 1};
  const int loc1 = (r2 + offx[ls1]) * 5 + (r3 + offy[ls1]);
  float lr1, cr1; step_rewards(cv1, loc1, cs1, &lr1, &cr1);

  out[0 * B + b] = clp0;  out[1 * B + b] = slp0;  out[2 * B + b] = llp0;
  out[3 * B + b] = clp1;  out[4 * B + b] = slp1;  out[5 * B + b] = llp1;
  out[6 * B + b] = alp1;
  out[7 * B + b]  = lr0;  out[8 * B + b]  = cr0;  out[9 * B + b]  = lr0;
  out[10 * B + b] = lr1;  out[11 * B + b] = cr1;  out[12 * B + b] = lr1;
  out[13 * B + b] = att_r;
}

// ---------------------------------------------------------------------------
extern "C" void kernel_launch(void* const* d_in, const int* in_sizes, int n_in,
                              void* d_out, int out_size, void* d_ws, size_t ws_size,
                              hipStream_t stream) {
  (void)n_in; (void)out_size; (void)ws_size;
  const int*   inst0   = (const int*)  d_in[0];
  const int*   inst1   = (const int*)  d_in[1];
  const int*   canvas0 = (const int*)  d_in[2];
  const int*   canvas1 = (const int*)  d_in[3];
  const int*   ref     = (const int*)  d_in[4];
  const float* embed   = (const float*)d_in[5];
  const float* W_ih    = (const float*)d_in[6];
  const float* W_hh    = (const float*)d_in[7];
  const float* b_ih    = (const float*)d_in[8];
  const float* b_hh    = (const float*)d_in[9];
  const float* Wc = (const float*)d_in[10]; const float* bc = (const float*)d_in[11];
  const float* Ws = (const float*)d_in[12]; const float* bs = (const float*)d_in[13];
  const float* Wl = (const float*)d_in[14]; const float* bl = (const float*)d_in[15];
  const float* Wr1 = (const float*)d_in[16]; const float* br1 = (const float*)d_in[17];
  const float* Wr2 = (const float*)d_in[18]; const float* br2 = (const float*)d_in[19];
  const float* Wp = (const float*)d_in[20]; const float* bp = (const float*)d_in[21];

  const int B = in_sizes[0] / SEQ;

  float* h0  = (float*)d_ws;
  float* h1  = h0 + (size_t)B * RNN;
  float* att = h1 + (size_t)B * RNN;
  float* out = (float*)d_out;

  const size_t lds_bytes = (size_t)GATES * RNN * 2 * sizeof(__bf16)  // W_ih + W_hh
                         + (size_t)GATES * sizeof(float)             // bias
                         + (size_t)64 * RNN * 2 * sizeof(__bf16);    // x + h
  lstm_kernel<<<dim3(B / 64, 2), 128, lds_bytes, stream>>>(
      inst0, inst1, embed, W_ih, W_hh, b_ih, b_hh, h0, h1, B);
  att_kernel<<<(B * 25 + 255) / 256, 256, 0, stream>>>(
      h1, canvas0, Wr1, br1, Wr2, br2, att, B);
  finalize_kernel<<<(B + 255) / 256, 256, 0, stream>>>(
      h0, h1, att, canvas0, canvas1, ref, Wc, bc, Ws, bs, Wl, bl, Wp, bp, out, B);
}
